// GroupedQueryAttention_678604833268
// MI455X (gfx1250) — matrix operand order repaired and checked by reading, not compiled, not run on hardware
//
#include <hip/hip_runtime.h>
#include <hip/hip_bf16.h>

#define T_    2048
#define D_    2560
#define H_    8
#define KV_   4
#define HD_   256
#define S_    8192
#define PREV_ 4096
#define SCALE_ 0.0625f   // 256^-0.5

typedef __attribute__((ext_vector_type(16))) __bf16 v16bf;
typedef __attribute__((ext_vector_type(8)))  __bf16 v8bf;
typedef __attribute__((ext_vector_type(8)))  float  v8f;

#define WMMA_BF16(a, b, c) \
  __builtin_amdgcn_wmma_f32_16x16x32_bf16(false, (a), false, (b), (short)0, (c), false, false)

// Async global->LDS copy (CDNA5): LDS address comes from the vdst VGPR; tracked by ASYNCcnt.
#define ASYNC_LOAD_B128(ldsoff_u32, gaddr_u64) \
  asm volatile("global_load_async_to_lds_b128 %0, %1, off" \
               :: "v"(ldsoff_u32), "v"(gaddr_u64) : "memory")
#define WAIT_ASYNC(n) asm volatile("s_wait_asynccnt %0" :: "n"(n) : "memory")

// ---- A fragment: 16x32 (MxK) bf16, row-major [M][K]; lane m%16 holds
// K chunks {kg*8..+8} and {16+kg*8..+8}, kg = lane>>4 (ISA 7.12.2). ----
static __device__ __forceinline__ v16bf load_a_frag(const __bf16* __restrict__ base, int ld) {
  const int lane = threadIdx.x & 31;
  const int r  = lane & 15;
  const int kb = (lane >> 4) << 3;
  const __bf16* p = base + (size_t)r * ld + kb;
  v8bf c0 = *(const v8bf*)(p);
  v8bf c1 = *(const v8bf*)(p + 16);
  v16bf f;
#pragma unroll
  for (int i = 0; i < 8; ++i) { f[i] = c0[i]; f[i + 8] = c1[i]; }
  return f;
}

// ---- B fragment: 32x16 (KxN) bf16 from row-major B^T [N][K]; lane n%16
// holds K chunks {kg*8..+8} and {16+kg*8..+8}, kg = lane>>4. ----
static __device__ __forceinline__ v16bf load_bt_frag(const __bf16* __restrict__ base, int ld) {
  const int lane = threadIdx.x & 31;
  const int n  = lane & 15;
  const int kb = (lane >> 4) << 3;
  const __bf16* p = base + (size_t)n * ld + kb;
  v8bf c0 = *(const v8bf*)(p);
  v8bf c1 = *(const v8bf*)(p + 16);
  v16bf f;
#pragma unroll
  for (int i = 0; i < 8; ++i) { f[i] = c0[i]; f[i + 8] = c1[i]; }
  return f;
}

// ---------------- elementwise fp32 -> bf16 ----------------
__global__ void conv_f32_bf16(const float* __restrict__ in, __bf16* __restrict__ out, size_t n) {
  size_t i = (size_t)blockIdx.x * blockDim.x + threadIdx.x;
  const size_t stride = (size_t)gridDim.x * blockDim.x;
  for (; i < n; i += stride) out[i] = (__bf16)in[i];
}

// k_cache/v_cache [s][g][d] f32 -> Kc [g][s][d] bf16, Vt [g][d][s] bf16
__global__ void conv_caches(const float* __restrict__ kc, const float* __restrict__ vc,
                            __bf16* __restrict__ Kc, __bf16* __restrict__ Vt) {
  const size_t n = (size_t)S_ * KV_ * HD_;
  size_t i = (size_t)blockIdx.x * blockDim.x + threadIdx.x;
  const size_t stride = (size_t)gridDim.x * blockDim.x;
  for (; i < n; i += stride) {
    const int d = (int)(i & (HD_ - 1));
    const int g = (int)((i >> 8) & (KV_ - 1));
    const int s = (int)(i >> 10);
    Kc[((size_t)g * S_ + s) * HD_ + d] = (__bf16)kc[i];
    Vt[((size_t)g * HD_ + d) * S_ + s] = (__bf16)vc[i];
  }
}

// ---------------- WMMA GEMM: C[M,N] f32 = A[M,K] * B[N,K]^T (bf16) ----------------
// 8 waves / block; wave owns a 32x64 tile. k-loop unrolled 2x into a ping/pong
// schedule: loads write directly into the carried fragment sets (no reg copies);
// each load set overlaps the other set's WMMAs. Requires K % 64 == 0.
__global__ __launch_bounds__(256)
void gemm_bf16_nt(const __bf16* __restrict__ A, const __bf16* __restrict__ B,
                  float* __restrict__ C, int M, int N, int K,
                  int lda, int ldb, int ldc) {
  const int wave = threadIdx.x >> 5;
  const int m0 = blockIdx.x * 32;
  const int n0 = (blockIdx.y * 8 + wave) * 64;
  if (n0 >= N || m0 >= M) return;

  const __bf16* Arow0 = A + (size_t)m0 * lda;
  const __bf16* Arow1 = A + (size_t)(m0 + 16) * lda;

  v8f acc[2][4] = {};
  v16bf a0A = load_a_frag(Arow0, lda);
  v16bf a1A = load_a_frag(Arow1, lda);
  v16bf bA[4], bB[4];
#pragma unroll
  for (int j = 0; j < 4; ++j) bA[j] = load_bt_frag(B + (size_t)(n0 + j * 16) * ldb, ldb);

  for (int k = 0; k < K; k += 64) {
    // stage set B (k+32): overlaps set-A WMMAs below
    v16bf a0B = load_a_frag(Arow0 + k + 32, lda);
    v16bf a1B = load_a_frag(Arow1 + k + 32, lda);
#pragma unroll
    for (int j = 0; j < 4; ++j)
      bB[j] = load_bt_frag(B + (size_t)(n0 + j * 16) * ldb + k + 32, ldb);
    // consume set A
#pragma unroll
    for (int j = 0; j < 4; ++j) {
      acc[0][j] = WMMA_BF16(a0A, bA[j], acc[0][j]);
      acc[1][j] = WMMA_BF16(a1A, bA[j], acc[1][j]);
    }
    // refill set A (k+64) directly into carried regs: overlaps set-B WMMAs
    if (k + 64 < K) {
      a0A = load_a_frag(Arow0 + k + 64, lda);
      a1A = load_a_frag(Arow1 + k + 64, lda);
#pragma unroll
      for (int j = 0; j < 4; ++j)
        bA[j] = load_bt_frag(B + (size_t)(n0 + j * 16) * ldb + k + 64, ldb);
    }
    // consume set B
#pragma unroll
    for (int j = 0; j < 4; ++j) {
      acc[0][j] = WMMA_BF16(a0B, bB[j], acc[0][j]);
      acc[1][j] = WMMA_BF16(a1B, bB[j], acc[1][j]);
    }
  }

  const int lane = threadIdx.x & 31;
  const int nn = lane & 15;
  const int mo = (lane >> 4) << 3;
#pragma unroll
  for (int mt = 0; mt < 2; ++mt)
#pragma unroll
    for (int j = 0; j < 4; ++j)
#pragma unroll
      for (int v = 0; v < 8; ++v)
        C[(size_t)(m0 + mt * 16 + mo + v) * ldc + (n0 + j * 16 + nn)] = acc[mt][j][v];
}

// ---------------- RMSNorm + RoPE + cache scatter ----------------
__global__ __launch_bounds__(256)
void normrope_kernel(const float* __restrict__ qkv, const float* __restrict__ qscale,
                     const float* __restrict__ kscale, const float* __restrict__ cosb,
                     const float* __restrict__ sinb,
                     __bf16* __restrict__ Qb, __bf16* __restrict__ Kc,
                     __bf16* __restrict__ Vt) {
  const int t = blockIdx.x;
  const int idx = blockIdx.y;
  const int d = threadIdx.x;

  if (idx >= H_ + KV_) {                       // V path: convert + transpose scatter
    const int g = idx - (H_ + KV_);
    const float v = qkv[(size_t)t * 4096 + 3072 + g * HD_ + d];
    Vt[((size_t)g * HD_ + d) * S_ + (PREV_ + t)] = (__bf16)v;
    return;
  }

  __shared__ float red[256];
  const bool isq = (idx < H_);
  const float* row = qkv + (size_t)t * 4096 + (isq ? idx * HD_ : 2048 + (idx - H_) * HD_);
  const float x = row[d];
  red[d] = x * x;
  __syncthreads();
  for (int st = 128; st > 0; st >>= 1) {
    if (d < st) red[d] += red[d + st];
    __syncthreads();
  }
  const float inv = rsqrtf(red[0] * (1.0f / 256.0f) + 1e-6f);
  __syncthreads();
  const float sc = isq ? qscale[d] : kscale[d];
  const float xn = x * inv * (1.0f + sc);
  red[d] = xn;
  __syncthreads();
  const float partner = (d < 128) ? -red[d + 128] : red[d - 128];
  const float r = xn * cosb[(size_t)t * HD_ + d] + partner * sinb[(size_t)t * HD_ + d];
  if (isq) {
    Qb[(size_t)t * (H_ * HD_) + idx * HD_ + d] = (__bf16)(r * SCALE_);
  } else {
    const int g = idx - H_;
    Kc[((size_t)g * S_ + (PREV_ + t)) * HD_ + d] = (__bf16)r;
  }
}

// ---------------- Flash attention ----------------
// Block = 128 threads = 4 waves = 4 consecutive 16-row q tiles of ONE head.
// K/V staged in LDS with async copies (K double-buffered, V waited after
// softmax). Q tile held in registers for the whole s-loop (8 A-fragments).
__global__ __launch_bounds__(128)
void attn_kernel(const __bf16* __restrict__ Qb, const __bf16* __restrict__ Kc,
                 const __bf16* __restrict__ Vt, __bf16* __restrict__ Ctx) {
  __shared__ __align__(16) __bf16 ktile[2][32 * HD_];   // [buf][s_local][d]
  __shared__ __align__(16) __bf16 vtile[HD_ * 32];      // [d][s_local]
  __shared__ __align__(16) __bf16 pbuf[4][16 * 32];     // per-wave P stage

  const int wave = threadIdx.x >> 5;
  const int lane = threadIdx.x & 31;
  const int ltid = threadIdx.x;
  const int t0b = blockIdx.x * 64;
  const int t0 = t0b + wave * 16;
  const int h = blockIdx.y;
  const int g = h >> 1;                                  // G = 2
  const int nn = lane & 15;
  const int mo = (lane >> 4) << 3;

  const __bf16* Qbase = Qb + (size_t)t0 * (H_ * HD_) + h * HD_;
  const __bf16* Kg = Kc + (size_t)g * S_ * HD_;
  const __bf16* Vg = Vt + (size_t)g * HD_ * S_;

  auto stage_k = [&](int s0, int buf) {
#pragma unroll
    for (int c = 0; c < 8; ++c) {
      const int chunk = ltid + c * 128;
      const int r = chunk >> 5, part = chunk & 31;
      const unsigned long long gp =
          (unsigned long long)(uintptr_t)(Kg + (size_t)(s0 + r) * HD_ + part * 8);
      const unsigned lp = (unsigned)(uintptr_t)&ktile[buf][r * HD_ + part * 8];
      ASYNC_LOAD_B128(lp, gp);
    }
  };
  auto stage_v = [&](int s0) {
#pragma unroll
    for (int c = 0; c < 8; ++c) {
      const int chunk = ltid + c * 128;
      const int d = chunk >> 2, part = chunk & 3;
      const unsigned long long gp =
          (unsigned long long)(uintptr_t)(Vg + (size_t)d * S_ + s0 + part * 8);
      const unsigned lp = (unsigned)(uintptr_t)&vtile[d * 32 + part * 8];
      ASYNC_LOAD_B128(lp, gp);
    }
  };

  // Q tile resident in registers for the whole kernel (8 x v16bf = 64 VGPRs)
  v16bf qf[8];
#pragma unroll
  for (int q8 = 0; q8 < 8; ++q8) qf[q8] = load_a_frag(Qbase + q8 * 32, H_ * HD_);

  v8f o[16] = {};
  float rmax[8], rsum[8];
#pragma unroll
  for (int v = 0; v < 8; ++v) { rmax[v] = -1e30f; rsum[v] = 0.0f; }

  const int n_tiles = (PREV_ + t0b + 64) / 32;
  stage_k(0, 0);

  for (int it = 0; it < n_tiles; ++it) {
    const int s0 = it * 32;
    const int buf = it & 1;
    const bool more = (it + 1) < n_tiles;

    __syncthreads();                      // prev PV reads of vtile done everywhere
    stage_v(s0);                          // 8 async
    if (more) stage_k(s0 + 32, buf ^ 1);  // 8 async
    if (more) WAIT_ASYNC(16); else WAIT_ASYNC(8);   // K(it) complete (in-order)
    __syncthreads();                      // K(it) resident for all waves

    // ---- scores: Q(16x256) x K^T -> 16x32, 4 accumulators break dep chains ----
    v8f sa0 = {}, sb0 = {}, sa1 = {}, sb1 = {};
#pragma unroll
    for (int i = 0; i < 4; ++i) {
      const int kk = i * 64;
      const v16bf k00 = load_bt_frag(&ktile[buf][kk], HD_);
      const v16bf k10 = load_bt_frag(&ktile[buf][16 * HD_ + kk], HD_);
      const v16bf k01 = load_bt_frag(&ktile[buf][kk + 32], HD_);
      const v16bf k11 = load_bt_frag(&ktile[buf][16 * HD_ + kk + 32], HD_);
      sa0 = WMMA_BF16(qf[2 * i],     k00, sa0);
      sa1 = WMMA_BF16(qf[2 * i],     k10, sa1);
      sb0 = WMMA_BF16(qf[2 * i + 1], k01, sb0);
      sb1 = WMMA_BF16(qf[2 * i + 1], k11, sb1);
    }
    const v8f sc0 = sa0 + sb0;
    const v8f sc1 = sa1 + sb1;

    // ---- causal mask + online softmax (row stats replicated per 16-lane half) ----
#pragma unroll
    for (int v = 0; v < 8; ++v) {
      const int lim = PREV_ + t0 + mo + v;
      float a0 = (s0 + nn      <= lim) ? sc0[v] : -1e30f;
      float a1 = (s0 + 16 + nn <= lim) ? sc1[v] : -1e30f;
      float mx = fmaxf(a0, a1);
#pragma unroll
      for (int off = 1; off < 16; off <<= 1) mx = fmaxf(mx, __shfl_xor(mx, off, 32));
      const float nm = fmaxf(rmax[v], mx);
      const float alpha = __expf(rmax[v] - nm);
      const float p0 = __expf(a0 - nm);
      const float p1 = __expf(a1 - nm);
      float ps = p0 + p1;
#pragma unroll
      for (int off = 1; off < 16; off <<= 1) ps += __shfl_xor(ps, off, 32);
      rsum[v] = rsum[v] * alpha + ps;
      rmax[v] = nm;
#pragma unroll
      for (int j = 0; j < 16; ++j) o[j][v] *= alpha;
      pbuf[wave][(mo + v) * 32 + nn]      = (__bf16)p0;
      pbuf[wave][(mo + v) * 32 + 16 + nn] = (__bf16)p1;
    }
    asm volatile("s_wait_dscnt 0" ::: "memory");
    v16bf pa;                                           // P as A fragment (wave-local LDS)
    {
      const int kb = (lane >> 4) << 3;
#pragma unroll
      for (int i = 0; i < 8; ++i) {
        pa[i]     = pbuf[wave][nn * 32 + kb + i];
        pa[i + 8] = pbuf[wave][nn * 32 + 16 + kb + i];
      }
    }

    if (more) WAIT_ASYNC(8); else WAIT_ASYNC(0);        // V(it) complete
    __syncthreads();                                    // V(it) resident for all waves
#pragma unroll
    for (int j = 0; j < 16; ++j) {                      // O += P(16x32) x V(32x16)
      const v16bf vb = load_bt_frag(&vtile[(j * 16) * 32], 32);
      o[j] = WMMA_BF16(pa, vb, o[j]);
    }
  }

#pragma unroll
  for (int j = 0; j < 16; ++j)
#pragma unroll
    for (int v = 0; v < 8; ++v) {
      const float r = o[j][v] / rsum[v];
      Ctx[(size_t)(t0 + mo + v) * (H_ * HD_) + h * HD_ + j * 16 + nn] = (__bf16)r;
    }
}

extern "C" void kernel_launch(void* const* d_in, const int* in_sizes, int n_in,
                              void* d_out, int out_size, void* d_ws, size_t ws_size,
                              hipStream_t stream) {
  (void)in_sizes; (void)n_in; (void)out_size; (void)ws_size;
  const float* x      = (const float*)d_in[0];
  const float* Wq     = (const float*)d_in[1];
  const float* Wk     = (const float*)d_in[2];
  const float* Wv     = (const float*)d_in[3];
  const float* Wo     = (const float*)d_in[4];
  const float* qs     = (const float*)d_in[5];
  const float* ks     = (const float*)d_in[6];
  const float* kcache = (const float*)d_in[7];
  const float* vcache = (const float*)d_in[8];
  const float* cosb   = (const float*)d_in[9];
  const float* sinb   = (const float*)d_in[10];

  char* ws = (char*)d_ws;
  __bf16* xb  = (__bf16*)(ws);                         // [2048][2560]
  __bf16* Wb  = (__bf16*)(ws + 10485760);              // [4096][2560]
  __bf16* Wob = (__bf16*)(ws + 31457280);              // [2560][2048]
  float*  qkv = (float*) (ws + 41943040);              // [2048][4096]
  __bf16* Qb  = (__bf16*)(ws + 75497472);              // [2048][2048]
  __bf16* Kc  = (__bf16*)(ws + 83886080);              // [4][8192][256]
  __bf16* Vt  = (__bf16*)(ws + 100663296);             // [4][256][8192]
  __bf16* Ctx = (__bf16*)(ws + 117440512);             // [2048][2048]

  conv_f32_bf16<<<2048, 256, 0, stream>>>(x,  xb,  (size_t)T_ * D_);
  conv_f32_bf16<<<2048, 256, 0, stream>>>(Wq, Wb,                      (size_t)2048 * D_);
  conv_f32_bf16<<<1024, 256, 0, stream>>>(Wk, Wb + (size_t)2048 * D_,  (size_t)1024 * D_);
  conv_f32_bf16<<<1024, 256, 0, stream>>>(Wv, Wb + (size_t)3072 * D_,  (size_t)1024 * D_);
  conv_f32_bf16<<<2048, 256, 0, stream>>>(Wo, Wob, (size_t)D_ * 2048);
  conv_caches  <<<4096, 256, 0, stream>>>(kcache, vcache, Kc, Vt);

  gemm_bf16_nt<<<dim3(64, 8), 256, 0, stream>>>(xb, Wb, qkv, 2048, 4096, 2560, 2560, 2560, 4096);
  normrope_kernel<<<dim3(2048, 16), 256, 0, stream>>>(qkv, qs, ks, cosb, sinb, Qb, Kc, Vt);
  attn_kernel<<<dim3(32, 8), 128, 0, stream>>>(Qb, Kc, Vt, Ctx);
  gemm_bf16_nt<<<dim3(64, 5), 256, 0, stream>>>(Ctx, Wob, (float*)d_out, 2048, 2560, 2048, 2048, 2048, 2560);
}
